// IEConvLayer_77266461655559
// MI455X (gfx1250) — compile-verified, matrix operand order checked by reading, not run on hardware
//
#include <hip/hip_runtime.h>
#include <hip/hip_bf16.h>

// ---------------------------------------------------------------------------
// IEConvLayer for MI455X (gfx1250): wave32 + v_wmma_f32_16x16x32_f16
// ---------------------------------------------------------------------------

#define NNODE   50000
#define NEDGE   800000
#define DIN     128
#define HDIM    16
#define DOUT    128
#define DEDGE   64
#define KHID    32
#define BN_EPS  1e-5f

#define EDGE_TILES       (NEDGE / 16)      // 50000
#define EDGE_TPW         10                // tiles per wave
#define EDGE_BLOCKS      (EDGE_TILES / (4 * EDGE_TPW))   // 1250, exact

typedef __attribute__((ext_vector_type(16))) _Float16 v16h;
typedef __attribute__((ext_vector_type(8)))  float    v8f;

__device__ __forceinline__ v8f wmma16(v16h a, v16h b, v8f c) {
    // D = A(16x32 f16) * B(32x16 f16) + C(16x16 f32)
    return __builtin_amdgcn_wmma_f32_16x16x32_f16(
        /*neg_a=*/false, a, /*neg_b=*/false, b,
        /*c_mod=*/(short)0, c, /*reuse_a=*/false, /*reuse_b=*/false);
}

// K index inside a 32-wide chunk for fragment half-element i of this lane.
// A: lane = 16*hi + m, holds row m, K = kb + (i<8 ? i : i+8), kb = 8*hi.
// B: lane = 16*hi + n, holds col n, same K pattern (ISA 7.12.2).
__device__ __forceinline__ int frag_k(int kb, int i) {
    return kb + (i < 8 ? i : i + 8);
}

// 16 contiguous halves (32B, aligned) -> v16h; lowers to 2x ds_load_b128.
__device__ __forceinline__ v16h lds_frag(const _Float16* p) {
    v16h r;
#pragma unroll
    for (int i = 0; i < 16; ++i) r[i] = p[i];
    return r;
}

// ---------------------------------------------------------------------------
// 1) Per-column sum / sum-of-squares (BN statistics), atomically accumulated.
// ---------------------------------------------------------------------------
__global__ void k_colstats(const float* __restrict__ x, int nrows, int ncols,
                           float* __restrict__ sums /* [2*ncols] */) {
    int c       = threadIdx.x % ncols;
    int rlane   = threadIdx.x / ncols;
    int rstride = blockDim.x / ncols;
    float s = 0.f, q = 0.f;
    for (int r = blockIdx.x * rstride + rlane; r < nrows;
         r += gridDim.x * rstride) {
        float v = x[(size_t)r * ncols + c];
        s += v; q += v * v;
    }
    atomicAdd(&sums[c], s);
    atomicAdd(&sums[ncols + c], q);
}

// sums -> scale/shift:  y = x*scale + shift  ==  g*(x-m)*rsqrt(v+eps)+b
__global__ void k_bn_finalize(const float* __restrict__ sums,
                              const float* __restrict__ g,
                              const float* __restrict__ b,
                              int ncols, float inv_n,
                              float* __restrict__ ss /* [2*ncols] */) {
    int c = blockIdx.x * blockDim.x + threadIdx.x;
    if (c >= ncols) return;
    float m  = sums[c] * inv_n;
    float v  = sums[ncols + c] * inv_n - m * m;
    float sc = g[c] * rsqrtf(v + BN_EPS);
    ss[c]         = sc;
    ss[ncols + c] = b[c] - m * sc;
}

// ---------------------------------------------------------------------------
// 2) Node kernel: h1 = relu(bn(x)) @ W1.T + b1      (N x 16), WMMA K=128.
//    One wave per 16-node tile (3125 tiles, exact).
// ---------------------------------------------------------------------------
__global__ void k_node(const float* __restrict__ x,
                       const float* __restrict__ bnX,   // [256] scale|shift
                       const float* __restrict__ W1,    // [16][128]
                       const float* __restrict__ b1,    // [16]
                       float* __restrict__ h1) {        // [N][16]
    const int lane = threadIdx.x & 31;
    const int wv   = threadIdx.x >> 5;
    const int tile = blockIdx.x * 4 + wv;
    if (tile >= NNODE / 16) return;
    const int row0 = tile * 16;
    const int m    = lane & 15;
    const int hi   = lane >> 4;
    const int kb   = hi * 8;

    const float* xr = x + (size_t)(row0 + m) * DIN;
    v8f acc = {};
#pragma unroll
    for (int kc = 0; kc < 4; ++kc) {
        v16h a, bfr;
#pragma unroll
        for (int i = 0; i < 16; ++i) {
            int k = kc * 32 + frag_k(kb, i);
            float v = fmaxf(xr[k] * bnX[k] + bnX[DIN + k], 0.f);
            a[i]   = (_Float16)v;
            bfr[i] = (_Float16)W1[m * DIN + k];   // B col n = lane&15 == m
        }
        acc = wmma16(a, bfr, acc);
    }
    const float bias = b1[m];                      // output col = lane&15
#pragma unroll
    for (int r = 0; r < 8; ++r)
        h1[(size_t)(row0 + r + hi * 8) * HDIM + m] = acc[r] + bias;
}

// ---------------------------------------------------------------------------
// 3) msg BN stats: sums over E of h1[src[e]][h] (per h) via LDS + atomics.
// ---------------------------------------------------------------------------
__global__ void k_msg_stats(const int* __restrict__ edge_list,
                            const float* __restrict__ h1,
                            float* __restrict__ sums /* [32] */) {
    __shared__ float s[32];
    if (threadIdx.x < 32) s[threadIdx.x] = 0.f;
    __syncthreads();
    for (int e = blockIdx.x * blockDim.x + threadIdx.x; e < NEDGE;
         e += gridDim.x * blockDim.x) {
        int src = edge_list[2 * e];
        const float* hr = h1 + (size_t)src * HDIM;
#pragma unroll
        for (int h = 0; h < HDIM; ++h) {
            float v = hr[h];
            atomicAdd(&s[h], v);
            atomicAdd(&s[HDIM + h], v * v);
        }
    }
    __syncthreads();
    if (threadIdx.x < 32) atomicAdd(&sums[threadIdx.x], s[threadIdx.x]);
}

// ---------------------------------------------------------------------------
// 4) Fused edge kernel. Persistent waves: each wave processes EDGE_TPW
//    16-edge tiles (1250 blocks x 4 waves x 10 tiles = 50000, exact).
//    Wk1/Wk2 are staged once per block into LDS, pre-swizzled into the
//    per-lane WMMA B-fragment layout (2x ds_load_b128 per fragment fetch).
//    Stage A: a = relu(edge_input @ Wk1.T + bk1)                 4 WMMAs
//    Stage B: 17 groups, fully unrolled: kern_j = a @ Wk2T_j + bk2, fused
//             with einsum('hk,k->h') in C-layout via shfl_xor butterflies
//             (no LDS, no barriers in the loop), then weighted scatter-add.
// ---------------------------------------------------------------------------
__global__ void k_edge(const float* __restrict__ edge_input, // [E][64]
                       const int*   __restrict__ edge_list,  // [E][2]
                       const float* __restrict__ edge_w,     // [E]
                       const float* __restrict__ Wk1,        // [32][64]
                       const float* __restrict__ bk1,        // [32]
                       const float* __restrict__ Wk2,        // [272][32]
                       const float* __restrict__ bk2,        // [272]
                       const float* __restrict__ h1,         // [N][16]
                       const float* __restrict__ bnMsg,      // [32]
                       float* __restrict__ update) {         // [N][16]
    // Pre-swizzled B fragments: [frag][lane][half]
    __shared__ __align__(32) _Float16 ldsW2[17][32][16];   // 17408 B
    __shared__ __align__(32) _Float16 ldsW1[4][32][16];    //  4096 B
    // Stage-A -> stage-B transpose bounce (per-wave region)
    __shared__ __align__(64) _Float16 aMat[4][16][32];     //  4096 B

    const int lane = threadIdx.x & 31;
    const int wv   = threadIdx.x >> 5;
    const int m    = lane & 15;     // A-row (edge) / B-col index
    const int hi   = lane >> 4;
    const int kb   = hi * 8;

    // ---- one-time cooperative staging of swizzled weights ------------------
    for (int idx = threadIdx.x; idx < 17 * 32 * 16; idx += blockDim.x) {
        int j  = idx >> 9;
        int ln = (idx >> 4) & 31;
        int i  = idx & 15;
        int k  = ((ln >> 4) << 3) + ((i < 8) ? i : i + 8);
        ldsW2[j][ln][i] =
            (_Float16)Wk2[(size_t)(j * 16 + (ln & 15)) * KHID + k];
    }
    for (int idx = threadIdx.x; idx < 4 * 32 * 16; idx += blockDim.x) {
        int f  = idx >> 9;             // f = kc*2 + nh
        int kc = f >> 1, nh = f & 1;
        int ln = (idx >> 4) & 31;
        int i  = idx & 15;
        int k  = kc * 32 + ((ln >> 4) << 3) + ((i < 8) ? i : i + 8);
        ldsW1[f][ln][i] =
            (_Float16)Wk1[(size_t)(nh * 16 + (ln & 15)) * DEDGE + k];
    }
    __syncthreads();

    // per-lane constants
    const float bb0 = bk1[m], bb1 = bk1[16 + m];
    const float scM = bnMsg[m], shM = bnMsg[HDIM + m];
    float bk2v[17];
#pragma unroll
    for (int j = 0; j < 17; ++j) bk2v[j] = bk2[j * 16 + m];

    const int wg = blockIdx.x * 4 + wv;            // 0..4999

    for (int it = 0; it < EDGE_TPW; ++it) {
        const int tile = wg * EDGE_TPW + it;
        const int e0   = tile * 16;

        // speculative prefetch of next tile's edge rows (global_prefetch_b8)
        {
            int pe = (e0 + 16 < NEDGE) ? (e0 + 16) : e0;
            __builtin_prefetch(&edge_input[(size_t)pe * DEDGE], 0, 0);
        }

        // ---- Stage A: edge_input (16x64) @ Wk1.T (64x32) -------------------
        v16h a0, a1;
        const float* erow = edge_input + (size_t)(e0 + m) * DEDGE;
#pragma unroll
        for (int i = 0; i < 16; ++i) {
            int k = frag_k(kb, i);
            a0[i] = (_Float16)erow[k];
            a1[i] = (_Float16)erow[32 + k];
        }
        v8f c0 = {}, c1 = {};
        c0 = wmma16(a0, lds_frag(&ldsW1[0][lane][0]), c0);   // kc0, n 0..15
        c0 = wmma16(a1, lds_frag(&ldsW1[2][lane][0]), c0);   // kc1, n 0..15
        c1 = wmma16(a0, lds_frag(&ldsW1[1][lane][0]), c1);   // kc0, n 16..31
        c1 = wmma16(a1, lds_frag(&ldsW1[3][lane][0]), c1);   // kc1, n 16..31

        // bias + relu -> per-wave LDS tile (16 edges x 32 hidden), f16
#pragma unroll
        for (int r = 0; r < 8; ++r) {
            aMat[wv][r + hi * 8][m]      = (_Float16)fmaxf(c0[r] + bb0, 0.f);
            aMat[wv][r + hi * 8][16 + m] = (_Float16)fmaxf(c1[r] + bb1, 0.f);
        }
        __syncthreads();   // uniform: all 4 waves run EDGE_TPW iterations

        // msg in C-layout: msgC[r] = relu(bn(h1[src[e]][k])), e=r+8*hi, k=m
        float msgC[8];
#pragma unroll
        for (int r = 0; r < 8; ++r) {
            int e = e0 + r + hi * 8;
            int s = edge_list[2 * e];
            msgC[r] = fmaxf(h1[(size_t)s * HDIM + m] * scM + shM, 0.f);
        }

        // Stage-B A fragment (16 edges x K=32) from LDS (2x ds_load_b128)
        v16h a2;
        const _Float16* ar = &aMat[wv][m][0];
#pragma unroll
        for (int i = 0; i < 16; ++i) a2[i] = ar[frag_k(kb, i)];

        // ---- Stage B + fused einsum (no barriers) --------------------------
        v8f cj = {};
        cj = wmma16(a2, lds_frag(&ldsW2[0][lane][0]), cj);   // bias group j=0
        float moC[8];
#pragma unroll
        for (int r = 0; r < 8; ++r) moC[r] = cj[r] + bk2v[0];

#pragma unroll
        for (int j = 1; j <= HDIM; ++j) {
            v8f ck = {};
            ck = wmma16(a2, lds_frag(&ldsW2[j][lane][0]), ck);
            float dot[8];
#pragma unroll
            for (int r = 0; r < 8; ++r)
                dot[r] = (ck[r] + bk2v[j]) * msgC[r];
            // butterfly sum over the 16-lane k-group (stays within half-wave)
#pragma unroll
            for (int sh = 1; sh <= 8; sh <<= 1) {
#pragma unroll
                for (int r = 0; r < 8; ++r)
                    dot[r] += __shfl_xor(dot[r], sh, 32);
            }
            const float sel = (m == (j - 1)) ? 1.f : 0.f;   // branchless
#pragma unroll
            for (int r = 0; r < 8; ++r) moC[r] += sel * dot[r];
        }

        // ---- weighted scatter-add straight out of C-layout -----------------
#pragma unroll
        for (int r = 0; r < 8; ++r) {
            int   e = e0 + r + hi * 8;
            int   d = edge_list[2 * e + 1];
            float w = edge_w[e];
            atomicAdd(&update[(size_t)d * HDIM + m], moC[r] * w);
        }
    }
}

// ---------------------------------------------------------------------------
// 5) Out kernel: out_pre = relu(bn(update)) @ W2.T + b2   (N x 128),
//    K padded 16 -> 32 with zeros. 8 WMMAs per 16-node tile.
// ---------------------------------------------------------------------------
__global__ void k_out(const float* __restrict__ update,  // [N][16]
                      const float* __restrict__ bnU,     // [32]
                      const float* __restrict__ W2,      // [128][16]
                      const float* __restrict__ bias2,   // [128]
                      float* __restrict__ outPre) {      // [N][128]
    const int lane = threadIdx.x & 31;
    const int wv   = threadIdx.x >> 5;
    const int tile = blockIdx.x * 4 + wv;
    if (tile >= NNODE / 16) return;
    const int row0 = tile * 16;
    const int m    = lane & 15;
    const int hi   = lane >> 4;
    const int kb   = hi * 8;

    v16h a;
    const float* ur = update + (size_t)(row0 + m) * HDIM;
#pragma unroll
    for (int i = 0; i < 16; ++i) {
        int k = frag_k(kb, i);
        float v = (k < HDIM)
                    ? fmaxf(ur[k] * bnU[k] + bnU[HDIM + k], 0.f) : 0.f;
        a[i] = (_Float16)v;
    }
#pragma unroll
    for (int g = 0; g < 8; ++g) {
        v16h bf;
#pragma unroll
        for (int i = 0; i < 16; ++i) {
            int k = frag_k(kb, i);
            bf[i] = (k < HDIM)
                      ? (_Float16)W2[(size_t)(g * 16 + m) * HDIM + k]
                      : (_Float16)0.f;
        }
        v8f c = {};
        c = wmma16(a, bf, c);
        float bb = bias2[g * 16 + m];
#pragma unroll
        for (int r = 0; r < 8; ++r)
            outPre[(size_t)(row0 + r + hi * 8) * DOUT + g * 16 + m] =
                c[r] + bb;
    }
}

// ---------------------------------------------------------------------------
// 6) Final elementwise BN apply.
// ---------------------------------------------------------------------------
__global__ void k_apply_bn(const float* __restrict__ pre,
                           const float* __restrict__ ss, int ncols,
                           size_t total, float* __restrict__ out) {
    for (size_t i = (size_t)blockIdx.x * blockDim.x + threadIdx.x; i < total;
         i += (size_t)gridDim.x * blockDim.x) {
        int c = (int)(i % ncols);
        out[i] = pre[i] * ss[c] + ss[ncols + c];
    }
}

// ---------------------------------------------------------------------------
extern "C" void kernel_launch(void* const* d_in, const int* in_sizes, int n_in,
                              void* d_out, int out_size, void* d_ws,
                              size_t ws_size, hipStream_t stream) {
    const float* x      = (const float*)d_in[0];
    const float* e_in   = (const float*)d_in[1];
    const int*   e_list = (const int*)  d_in[2];
    const float* e_w    = (const float*)d_in[3];
    const float* g_in   = (const float*)d_in[4];
    const float* b_in   = (const float*)d_in[5];
    const float* W1     = (const float*)d_in[6];
    const float* b1     = (const float*)d_in[7];
    const float* Wk1    = (const float*)d_in[8];
    const float* bk1    = (const float*)d_in[9];
    const float* Wk2    = (const float*)d_in[10];
    const float* bk2    = (const float*)d_in[11];
    const float* g_msg  = (const float*)d_in[12];
    const float* b_msg  = (const float*)d_in[13];
    const float* g_upd  = (const float*)d_in[14];
    const float* b_upd  = (const float*)d_in[15];
    const float* W2     = (const float*)d_in[16];
    const float* b2v    = (const float*)d_in[17];
    const float* g_out  = (const float*)d_in[18];
    const float* b_out  = (const float*)d_in[19];
    float* out = (float*)d_out;

    // ----- workspace layout (floats) -----
    float* ws      = (float*)d_ws;
    float* statX   = ws;              // 256  (sum | sumsq)
    float* bnX     = statX   + 256;   // 256  (scale | shift)
    float* statMsg = bnX     + 256;   // 32
    float* bnMsg   = statMsg + 32;    // 32
    float* statUpd = bnMsg   + 32;    // 32
    float* bnUpd   = statUpd + 32;    // 32
    float* statOut = bnUpd   + 32;    // 256
    float* bnOut   = statOut + 256;   // 256
    float* h1      = bnOut   + 256;                 // N*16
    float* update  = h1      + (size_t)NNODE * 16;  // N*16
    float* outPre  = update  + (size_t)NNODE * 16;  // N*128

    // zero the atomic accumulators (stats block + update)
    hipMemsetAsync(statX, 0, 1152 * sizeof(float), stream);
    hipMemsetAsync(update, 0, (size_t)NNODE * 16 * sizeof(float), stream);

    // ----- input BN -----
    k_colstats<<<512, 256, 0, stream>>>(x, NNODE, DIN, statX);
    k_bn_finalize<<<1, 128, 0, stream>>>(statX, g_in, b_in, DIN,
                                         1.0f / NNODE, bnX);
    // ----- per-node h1 (WMMA) -----
    k_node<<<(NNODE / 16 + 3) / 4, 128, 0, stream>>>(x, bnX, W1, b1, h1);

    // ----- msg BN -----
    k_msg_stats<<<512, 256, 0, stream>>>(e_list, h1, statMsg);
    k_bn_finalize<<<1, 128, 0, stream>>>(statMsg, g_msg, b_msg, HDIM,
                                         1.0f / NEDGE, bnMsg);

    // ----- fused edge kernel (WMMA), persistent waves -----
    k_edge<<<EDGE_BLOCKS, 128, 0, stream>>>(e_in, e_list, e_w, Wk1, bk1,
                                            Wk2, bk2, h1, bnMsg, update);

    // ----- update BN -----
    k_colstats<<<512, 256, 0, stream>>>(update, NNODE, HDIM, statUpd);
    k_bn_finalize<<<1, 128, 0, stream>>>(statUpd, g_upd, b_upd, HDIM,
                                         1.0f / NNODE, bnUpd);

    // ----- output projection (WMMA) -----
    k_out<<<(NNODE / 16 + 3) / 4, 128, 0, stream>>>(update, bnUpd, W2, b2v,
                                                    outPre);

    // ----- output BN -----
    k_colstats<<<512, 256, 0, stream>>>(outPre, NNODE, DOUT, statOut);
    k_bn_finalize<<<1, 128, 0, stream>>>(statOut, g_out, b_out, DOUT,
                                         1.0f / NNODE, bnOut);
    k_apply_bn<<<1024, 256, 0, stream>>>(outPre, bnOut, DOUT,
                                         (size_t)NNODE * DOUT, out);
}
